// OCAB_19189913879196
// MI455X (gfx1250) — compile-verified, matrix operand order checked by reading, not run on hardware
//
#include <hip/hip_runtime.h>
#include <hip/hip_bf16.h>
#include <math.h>

typedef __attribute__((ext_vector_type(16))) __bf16 v16bf;
typedef __attribute__((ext_vector_type(8)))  float  v8f;

#define C_DIM   180
#define C_PAD   192
#define C2_DIM  360
#define C2_PAD  384
#define NHEAD   6
#define HDIM    30
#define HPAD    32
#define WSZ     16
#define OWSZ    24
#define PADW    4
#define NQ      256
#define NK      576
#define NWIN    256
#define NTOK    65536
#define IMG     256

static __device__ __forceinline__ unsigned short f2bf(float f) {
  unsigned u = __builtin_bit_cast(unsigned, f);
  u += 0x7FFFu + ((u >> 16) & 1u);           // round-to-nearest-even
  return (unsigned short)(u >> 16);
}

union FragU { uint4 q[2]; v16bf v; };

// 16-bit A/B^T fragment: lane half hf selects K sub-chunks {hf*8..hf*8+7, hf*8+16..hf*8+23}
// within a 32-element K chunk of this lane's row (ISA 7.12.2 16-bit A 16x32 layout).
static __device__ __forceinline__ v16bf load_frag(const unsigned short* row, int hf) {
  FragU u;
  u.q[0] = *(const uint4*)(row + hf * 8);
  u.q[1] = *(const uint4*)(row + hf * 8 + 16);
  return u.v;
}
static __device__ __forceinline__ v16bf zero_frag() {
  FragU u;
  u.q[0] = make_uint4(0u, 0u, 0u, 0u);
  u.q[1] = make_uint4(0u, 0u, 0u, 0u);
  return u.v;
}

// ---------------- weight prep: transpose + pad + bf16 ----------------
__global__ void prep_weights(const float* qw, const float* kvw, const float* pw,
                             unsigned short* qwT, unsigned short* kvT, unsigned short* pT) {
  int idx = blockIdx.x * blockDim.x + threadIdx.x;
  const int T0 = C_PAD * C_PAD, T1 = C2_PAD * C_PAD, total = T0 + T1 + T0;
  for (; idx < total; idx += gridDim.x * blockDim.x) {
    int i = idx;
    if (i < T0) {
      int n = i / C_PAD, k = i % C_PAD;
      float v = (n < C_DIM && k < C_DIM) ? qw[k * C_DIM + n] : 0.f;
      qwT[n * C_PAD + k] = f2bf(v);
    } else if (i < T0 + T1) {
      i -= T0;
      int n = i / C_PAD, k = i % C_PAD;
      float v = (n < C2_DIM && k < C_DIM) ? kvw[k * C2_DIM + n] : 0.f;
      kvT[n * C_PAD + k] = f2bf(v);
    } else {
      i -= T0 + T1;
      int n = i / C_PAD, k = i % C_PAD;
      float v = (n < C_DIM && k < C_DIM) ? pw[k * C_DIM + n] : 0.f;
      pT[n * C_PAD + k] = f2bf(v);
    }
  }
}

// ---------------- LayerNorm: one wave per token, fp32 -> padded bf16 ----------------
__global__ void layernorm_k(const float* x, const float* nw, const float* nb,
                            unsigned short* xn) {
  int wave = threadIdx.x >> 5, lane = threadIdx.x & 31;
  int t = blockIdx.x * 8 + wave;
  const float* xr = x + (size_t)t * C_DIM;
  float vals[6], s = 0.f, s2 = 0.f;
#pragma unroll
  for (int i = 0; i < 6; i++) {
    int c = i * 32 + lane;
    float v = (c < C_DIM) ? xr[c] : 0.f;
    vals[i] = v; s += v; s2 += v * v;
  }
#pragma unroll
  for (int m = 16; m >= 1; m >>= 1) { s += __shfl_xor(s, m); s2 += __shfl_xor(s2, m); }
  float mu  = s / (float)C_DIM;
  float var = s2 / (float)C_DIM - mu * mu;
  float inv = rsqrtf(var + 1e-5f);
  unsigned short* o = xn + (size_t)t * C_PAD;
#pragma unroll
  for (int i = 0; i < 6; i++) {
    int c = i * 32 + lane;
    o[c] = (c < C_DIM) ? f2bf((vals[i] - mu) * inv * nw[c] + nb[c]) : (unsigned short)0;
  }
}

// epilogue helper: store one 16x16 f32 C tile into qb[w][h][q][32] with bias+scale
static __device__ __forceinline__ void store_q_tile(const v8f& acc, int mbase, int n,
                                                    int hf, const float* qbias,
                                                    unsigned short* qb) {
  if (n >= C_DIM) return;
  int h = n / HDIM, d = n % HDIM;
  float bias = qbias[n];
  const float scale = 0.18257418583505536f;               // 30^-0.5
#pragma unroll
  for (int rr = 0; rr < 8; rr++) {
    int m = mbase + hf * 8 + rr;
    int ww = m >> 8, pp = m & 255;
    qb[(((size_t)(ww * NHEAD + h)) * NQ + pp) * HPAD + d] = f2bf((acc[rr] + bias) * scale);
  }
}

// ---------------- Q GEMM (2x2 register-blocked): xn @ q_w + b -> qb[w][h][q][32] ----------------
__global__ void q_gemm(const unsigned short* xn, const unsigned short* qwT,
                       const float* qbias, unsigned short* qb) {
  const int NP = C_PAD / 32;                              // 6 N tile-pairs
  int wid  = blockIdx.x * (blockDim.x >> 5) + (threadIdx.x >> 5);
  int lane = threadIdx.x & 31, hf = lane >> 4, lm = lane & 15;
  int mp = wid / NP, np = wid % NP;                       // 32x32 output block
  // two window-ordered A rows -> tokens
  const unsigned short* arow[2];
#pragma unroll
  for (int t = 0; t < 2; t++) {
    int r = mp * 32 + t * 16 + lm;
    int w = r >> 8, p = r & 255;
    int token = (((w >> 4) * WSZ + (p >> 4)) << 8) + ((w & 15) * WSZ + (p & 15));
    arow[t] = xn + (size_t)token * C_PAD;
  }
  const unsigned short* brow0 = qwT + (size_t)(np * 32 + lm) * C_PAD;
  const unsigned short* brow1 = brow0 + (size_t)16 * C_PAD;
  v8f a00 = {}, a01 = {}, a10 = {}, a11 = {};
#pragma unroll
  for (int k0 = 0; k0 < C_PAD; k0 += 32) {
    v16bf fa0 = load_frag(arow[0] + k0, hf);
    v16bf fa1 = load_frag(arow[1] + k0, hf);
    v16bf fb0 = load_frag(brow0 + k0, hf);
    v16bf fb1 = load_frag(brow1 + k0, hf);
    a00 = __builtin_amdgcn_wmma_f32_16x16x32_bf16(false, fa0, false, fb0, (short)0, a00, false, false);
    a01 = __builtin_amdgcn_wmma_f32_16x16x32_bf16(false, fa0, false, fb1, (short)0, a01, false, false);
    a10 = __builtin_amdgcn_wmma_f32_16x16x32_bf16(false, fa1, false, fb0, (short)0, a10, false, false);
    a11 = __builtin_amdgcn_wmma_f32_16x16x32_bf16(false, fa1, false, fb1, (short)0, a11, false, false);
  }
  int n0 = np * 32 + lm, n1 = n0 + 16;
  store_q_tile(a00, mp * 32,      n0, hf, qbias, qb);
  store_q_tile(a01, mp * 32,      n1, hf, qbias, qb);
  store_q_tile(a10, mp * 32 + 16, n0, hf, qbias, qb);
  store_q_tile(a11, mp * 32 + 16, n1, hf, qbias, qb);
}

// epilogue helper for KV: n<180 -> K tile, 180<=n<360 -> V^T tile
static __device__ __forceinline__ void store_kv_tile(const v8f& acc, int mbase, int n,
                                                     int hf, const float* kvbias,
                                                     unsigned short* kb, unsigned short* vT) {
  if (n >= C2_DIM) return;
  float bias = kvbias[n];
#pragma unroll
  for (int rr = 0; rr < 8; rr++) {
    int m = mbase + hf * 8 + rr;
    int ww = m / NK, pp = m % NK;
    float v = acc[rr] + bias;
    if (n < C_DIM) {
      int h = n / HDIM, d = n % HDIM;
      kb[(((size_t)(ww * NHEAD + h)) * NK + pp) * HPAD + d] = f2bf(v);
    } else {
      int c = n - C_DIM, h = c / HDIM, d = c % HDIM;
      vT[(((size_t)(ww * NHEAD + h)) * HPAD + d) * NK + pp] = f2bf(v);
    }
  }
}

// ---------------- KV GEMM (2x2 blocked): overlap-gathered xn @ kv_w + b ----------------
__global__ void kv_gemm(const unsigned short* xn, const unsigned short* kvT,
                        const float* kvbias, unsigned short* kb, unsigned short* vT) {
  const int NP = C2_PAD / 32;                             // 12 N tile-pairs
  int wid  = blockIdx.x * (blockDim.x >> 5) + (threadIdx.x >> 5);
  int lane = threadIdx.x & 31, hf = lane >> 4, lm = lane & 15;
  int mp = wid / NP, np = wid % NP;
  const unsigned short* arow[2];
  bool valid[2];
#pragma unroll
  for (int t = 0; t < 2; t++) {
    int r = mp * 32 + t * 16 + lm;
    int w = r / NK, p = r % NK;
    int py = p / OWSZ, px = p % OWSZ;
    int gy = (w >> 4) * WSZ + py - PADW;
    int gx = (w & 15) * WSZ + px - PADW;
    valid[t] = (gy >= 0) && (gy < IMG) && (gx >= 0) && (gx < IMG);
    arow[t] = xn + (size_t)(valid[t] ? (gy * IMG + gx) : 0) * C_PAD;
  }
  const unsigned short* brow0 = kvT + (size_t)(np * 32 + lm) * C_PAD;
  const unsigned short* brow1 = brow0 + (size_t)16 * C_PAD;
  v8f a00 = {}, a01 = {}, a10 = {}, a11 = {};
#pragma unroll
  for (int k0 = 0; k0 < C_PAD; k0 += 32) {
    v16bf fa0 = valid[0] ? load_frag(arow[0] + k0, hf) : zero_frag();
    v16bf fa1 = valid[1] ? load_frag(arow[1] + k0, hf) : zero_frag();
    v16bf fb0 = load_frag(brow0 + k0, hf);
    v16bf fb1 = load_frag(brow1 + k0, hf);
    a00 = __builtin_amdgcn_wmma_f32_16x16x32_bf16(false, fa0, false, fb0, (short)0, a00, false, false);
    a01 = __builtin_amdgcn_wmma_f32_16x16x32_bf16(false, fa0, false, fb1, (short)0, a01, false, false);
    a10 = __builtin_amdgcn_wmma_f32_16x16x32_bf16(false, fa1, false, fb0, (short)0, a10, false, false);
    a11 = __builtin_amdgcn_wmma_f32_16x16x32_bf16(false, fa1, false, fb1, (short)0, a11, false, false);
  }
  int n0 = np * 32 + lm, n1 = n0 + 16;
  store_kv_tile(a00, mp * 32,      n0, hf, kvbias, kb, vT);
  store_kv_tile(a01, mp * 32,      n1, hf, kvbias, kb, vT);
  store_kv_tile(a10, mp * 32 + 16, n0, hf, kvbias, kb, vT);
  store_kv_tile(a11, mp * 32 + 16, n1, hf, kvbias, kb, vT);
}

// ---------------- Flash attention per (window, head): 256 q x 576 k x 30 d ----------------
__global__ void __launch_bounds__(512) attn_k(const unsigned short* qb,
                                              const unsigned short* kb,
                                              const unsigned short* vT,
                                              unsigned short* attnout) {
  __shared__ unsigned short Plds[16][16 * 64];            // per-wave P tile, 32 KB
  int wh = blockIdx.x;
  int w = wh / NHEAD, h = wh % NHEAD;
  int wave = threadIdx.x >> 5, lane = threadIdx.x & 31;
  int hf = lane >> 4, lm = lane & 15;
  int qbase = wave * 16;
  const unsigned short* qrow = qb + (((size_t)(w * NHEAD + h)) * NQ + qbase + lm) * HPAD;
  v16bf aq = load_frag(qrow, hf);                         // 16 q rows x 32 d, kept resident
  const unsigned short* kptr = kb + ((size_t)(w * NHEAD + h)) * NK * HPAD;
  const unsigned short* vptr = vT + ((size_t)(w * NHEAD + h)) * HPAD * NK;
  float m8[8], l8[8];
#pragma unroll
  for (int i = 0; i < 8; i++) { m8[i] = -1e30f; l8[i] = 0.f; }
  v8f o0 = {}, o1 = {};
  for (int kc = 0; kc < NK; kc += 64) {
    v8f s[4];
#pragma unroll
    for (int j = 0; j < 4; j++) {
      v16bf bk = load_frag(kptr + (size_t)(kc + j * 16 + lm) * HPAD, hf);
      v8f z = {};
      s[j] = __builtin_amdgcn_wmma_f32_16x16x32_bf16(false, aq, false, bk, (short)0, z, false, false);
    }
#pragma unroll
    for (int rr = 0; rr < 8; rr++) {
      float mx = fmaxf(fmaxf(s[0][rr], s[1][rr]), fmaxf(s[2][rr], s[3][rr]));
      mx = fmaxf(mx, __shfl_xor(mx, 1)); mx = fmaxf(mx, __shfl_xor(mx, 2));
      mx = fmaxf(mx, __shfl_xor(mx, 4)); mx = fmaxf(mx, __shfl_xor(mx, 8));
      float mn = fmaxf(m8[rr], mx);
      float alpha = __expf(m8[rr] - mn);
      float rs = 0.f;
#pragma unroll
      for (int j = 0; j < 4; j++) {
        float pv = __expf(s[j][rr] - mn);
        s[j][rr] = pv; rs += pv;
      }
      rs += __shfl_xor(rs, 1); rs += __shfl_xor(rs, 2);
      rs += __shfl_xor(rs, 4); rs += __shfl_xor(rs, 8);
      l8[rr] = l8[rr] * alpha + rs;
      o0[rr] *= alpha; o1[rr] *= alpha;
      m8[rr] = mn;
#pragma unroll
      for (int j = 0; j < 4; j++)
        Plds[wave][(hf * 8 + rr) * 64 + j * 16 + lm] = f2bf(s[j][rr]);
    }
    // P buffer is private per wave; LDS ops are in-order per wave (DScnt), so a
    // compiler fence + explicit dscnt wait replaces a block-wide barrier.
    __builtin_amdgcn_wave_barrier();
    asm volatile("s_wait_dscnt 0x0" ::: "memory");
#pragma unroll
    for (int ss = 0; ss < 2; ss++) {
      v16bf ap  = load_frag(&Plds[wave][lm * 64 + ss * 32], hf);
      v16bf bv0 = load_frag(vptr + (size_t)lm * NK + kc + ss * 32, hf);
      o0 = __builtin_amdgcn_wmma_f32_16x16x32_bf16(false, ap, false, bv0, (short)0, o0, false, false);
      v16bf bv1 = load_frag(vptr + (size_t)(16 + lm) * NK + kc + ss * 32, hf);
      o1 = __builtin_amdgcn_wmma_f32_16x16x32_bf16(false, ap, false, bv1, (short)0, o1, false, false);
    }
    __builtin_amdgcn_wave_barrier();
  }
#pragma unroll
  for (int rr = 0; rr < 8; rr++) {
    float inv = 1.f / l8[rr];
    int p = qbase + hf * 8 + rr;
    unsigned short* orow = attnout + (size_t)(w * NQ + p) * C_PAD + h * HDIM;
    orow[lm] = f2bf(o0[rr] * inv);
    if (lm < HDIM - 16) orow[16 + lm] = f2bf(o1[rr] * inv);
  }
}

// epilogue helper: proj tile + window-reverse + residual, fp32 out
static __device__ __forceinline__ void store_p_tile(const v8f& acc, int mbase, int n,
                                                    int hf, const float* pbias,
                                                    const float* x, float* out) {
  if (n >= C_DIM) return;
  float bias = pbias[n];
#pragma unroll
  for (int rr = 0; rr < 8; rr++) {
    int m = mbase + hf * 8 + rr;
    int ww = m >> 8, pp = m & 255;
    int token = (((ww >> 4) * WSZ + (pp >> 4)) << 8) + ((ww & 15) * WSZ + (pp & 15));
    out[(size_t)token * C_DIM + n] = x[(size_t)token * C_DIM + n] + bias + acc[rr];
  }
}

// ---------------- proj GEMM (2x2 blocked) + window-reverse + residual ----------------
__global__ void proj_gemm(const unsigned short* attnout, const unsigned short* pT,
                          const float* pbias, const float* x, float* out) {
  const int NP = C_PAD / 32;                              // 6 N tile-pairs
  int wid  = blockIdx.x * (blockDim.x >> 5) + (threadIdx.x >> 5);
  int lane = threadIdx.x & 31, hf = lane >> 4, lm = lane & 15;
  int mp = wid / NP, np = wid % NP;
  const unsigned short* arow0 = attnout + (size_t)(mp * 32 + lm) * C_PAD;
  const unsigned short* arow1 = arow0 + (size_t)16 * C_PAD;
  const unsigned short* brow0 = pT + (size_t)(np * 32 + lm) * C_PAD;
  const unsigned short* brow1 = brow0 + (size_t)16 * C_PAD;
  v8f a00 = {}, a01 = {}, a10 = {}, a11 = {};
#pragma unroll
  for (int k0 = 0; k0 < C_PAD; k0 += 32) {
    v16bf fa0 = load_frag(arow0 + k0, hf);
    v16bf fa1 = load_frag(arow1 + k0, hf);
    v16bf fb0 = load_frag(brow0 + k0, hf);
    v16bf fb1 = load_frag(brow1 + k0, hf);
    a00 = __builtin_amdgcn_wmma_f32_16x16x32_bf16(false, fa0, false, fb0, (short)0, a00, false, false);
    a01 = __builtin_amdgcn_wmma_f32_16x16x32_bf16(false, fa0, false, fb1, (short)0, a01, false, false);
    a10 = __builtin_amdgcn_wmma_f32_16x16x32_bf16(false, fa1, false, fb0, (short)0, a10, false, false);
    a11 = __builtin_amdgcn_wmma_f32_16x16x32_bf16(false, fa1, false, fb1, (short)0, a11, false, false);
  }
  int n0 = np * 32 + lm, n1 = n0 + 16;
  store_p_tile(a00, mp * 32,      n0, hf, pbias, x, out);
  store_p_tile(a01, mp * 32,      n1, hf, pbias, x, out);
  store_p_tile(a10, mp * 32 + 16, n0, hf, pbias, x, out);
  store_p_tile(a11, mp * 32 + 16, n1, hf, pbias, x, out);
}

extern "C" void kernel_launch(void* const* d_in, const int* in_sizes, int n_in,
                              void* d_out, int out_size, void* d_ws, size_t ws_size,
                              hipStream_t stream) {
  (void)in_sizes; (void)n_in; (void)out_size; (void)ws_size;
  const float* x      = (const float*)d_in[0];
  const float* norm_w = (const float*)d_in[1];
  const float* norm_b = (const float*)d_in[2];
  const float* q_w    = (const float*)d_in[3];
  const float* q_b    = (const float*)d_in[4];
  const float* kv_w   = (const float*)d_in[5];
  const float* kv_b   = (const float*)d_in[6];
  const float* proj_w = (const float*)d_in[7];
  const float* proj_b = (const float*)d_in[8];
  float* out = (float*)d_out;

  char* ws = (char*)d_ws;
  size_t off = 0;
  auto alloc = [&](size_t bytes) -> void* {
    void* p = ws + off; off += (bytes + 255) & ~(size_t)255; return p;
  };
  unsigned short* xn   = (unsigned short*)alloc((size_t)NTOK * C_PAD * 2);  // reused as attnout
  unsigned short* qwT  = (unsigned short*)alloc((size_t)C_PAD * C_PAD * 2);
  unsigned short* kvT  = (unsigned short*)alloc((size_t)C2_PAD * C_PAD * 2);
  unsigned short* pT   = (unsigned short*)alloc((size_t)C_PAD * C_PAD * 2);
  unsigned short* qbuf = (unsigned short*)alloc((size_t)NWIN * NHEAD * NQ * HPAD * 2);
  unsigned short* kbuf = (unsigned short*)alloc((size_t)NWIN * NHEAD * NK * HPAD * 2);
  unsigned short* vbuf = (unsigned short*)alloc((size_t)NWIN * NHEAD * HPAD * NK * 2);

  // zero the d=30,31 pad lanes of q/k so the 16x16x32 WMMA K-padding is exact
  hipMemsetAsync(qbuf, 0, (size_t)NWIN * NHEAD * NQ * HPAD * 2, stream);
  hipMemsetAsync(kbuf, 0, (size_t)NWIN * NHEAD * NK * HPAD * 2, stream);

  prep_weights<<<64, 256, 0, stream>>>(q_w, kv_w, proj_w, qwT, kvT, pT);
  layernorm_k<<<NTOK / 8, 256, 0, stream>>>(x, norm_w, norm_b, xn);
  // q: 4096 M tiles -> 2048 pairs, 6 N pairs; 8 waves/block
  q_gemm<<<(2048 * 6) / 8, 256, 0, stream>>>(xn, qwT, q_b, qbuf);
  // kv: 9216 M tiles -> 4608 pairs, 12 N pairs
  kv_gemm<<<(4608 * 12) / 8, 256, 0, stream>>>(xn, kvT, kv_b, kbuf, vbuf);
  attn_k<<<NWIN * NHEAD, 512, 0, stream>>>(qbuf, kbuf, vbuf, xn /* attnout reuse */);
  proj_gemm<<<(2048 * 6) / 8, 256, 0, stream>>>(xn, pT, proj_b, x, out);
}
